// T5LayerSelfAttention_87247965650972
// MI455X (gfx1250) — matrix operand order repaired and checked by reading, not compiled, not run on hardware
//
#include <hip/hip_runtime.h>

// ---------------------------------------------------------------------------
// T5 self-attention layer for gfx1250 (MI455X), bf16 WMMA pipeline.
// B=2, S=2048, D=1024, H=16, d_kv=64, NUM_BUCKETS=32, MAX_DISTANCE=128.
// GEMMs: 128x64 block tile, 2x2 WMMA tiles per wave, LDS double-buffered,
// global_prefetch for k+2 tiles. Attention: flash-style, online softmax.
// ---------------------------------------------------------------------------

typedef __attribute__((ext_vector_type(16))) __bf16 v16bf;
typedef __attribute__((ext_vector_type(8)))  float  v8f;

union FragU { v16bf v; unsigned u[8]; };
union U8x16 { uint4 u; unsigned short h[8]; };

__device__ __forceinline__ unsigned short f2bfu(float f) {
  unsigned u = __builtin_bit_cast(unsigned, f);
  return (unsigned short)((u + 0x7FFFu + ((u >> 16) & 1u)) >> 16);
}

// A-fragment (16x32, M x K), per ISA 16-bit A layout:
// lane m = lane&15; lanes<16 -> K-offset 0, lanes>=16 -> K-offset 8.
// VGPR j (j<4): K = koff+2j,+1 ; VGPR 4+j: K = 16+koff+2j,+1.
__device__ __forceinline__ v16bf load_frag_a(const unsigned short* p, int ld) {
  int lane = threadIdx.x & 31;
  int m = lane & 15;
  int koff = (lane < 16) ? 0 : 8;
  const unsigned* row = (const unsigned*)(p + (size_t)m * ld);
  FragU f;
#pragma unroll
  for (int j = 0; j < 4; ++j) {
    f.u[j]     = row[(koff >> 1) + j];
    f.u[4 + j] = row[((16 + koff) >> 1) + j];
  }
  return f.v;
}

// B-fragment (32x16, K x N) read from LDS stored as Bt[n][k] (k contiguous):
// lane n = lane&15; lanes<16 -> K-offset 0, lanes>=16 -> K-offset 8.
// VGPR j (j<4): K = koff+2j,+1 ; VGPR 4+j: K = 16+koff+2j,+1.
__device__ __forceinline__ v16bf load_frag_b(const unsigned short* p, int ld) {
  int lane = threadIdx.x & 31;
  int nn = lane & 15;
  int koff = (lane < 16) ? 0 : 8;
  const unsigned* row = (const unsigned*)(p + (size_t)nn * ld);
  FragU f;
#pragma unroll
  for (int j = 0; j < 4; ++j) {
    f.u[j]     = row[(koff >> 1) + j];
    f.u[4 + j] = row[((16 + koff) >> 1) + j];
  }
  return f.v;
}

#define WMMA_BF16(A, B, C) \
  __builtin_amdgcn_wmma_f32_16x16x32_bf16(false, (A), false, (B), (short)0, (C), false, false)

// ---------------------------------------------------------------------------
// fp32 -> bf16 elementwise conversion (weights)
// ---------------------------------------------------------------------------
__global__ void cvt_bf16_kernel(const float* __restrict__ src,
                                unsigned short* __restrict__ dst, int n) {
  int i = blockIdx.x * blockDim.x + threadIdx.x;
  if (i < n) dst[i] = f2bfu(src[i]);
}

// ---------------------------------------------------------------------------
// T5 RMS-norm: one 256-thread block per row of [4096 x 1024], bf16 output.
// ---------------------------------------------------------------------------
__global__ __launch_bounds__(256) void rmsnorm_kernel(
    const float* __restrict__ x, const float* __restrict__ g,
    unsigned short* __restrict__ out) {
  int row = blockIdx.x;
  int tid = threadIdx.x;
  const float* xr = x + (size_t)row * 1024;
  float4 v = *(const float4*)(xr + tid * 4);
  float s = v.x * v.x + v.y * v.y + v.z * v.z + v.w * v.w;
#pragma unroll
  for (int off = 16; off >= 1; off >>= 1) s += __shfl_xor(s, off, 32);
  __shared__ float ps[8];
  if ((tid & 31) == 0) ps[tid >> 5] = s;
  __syncthreads();
  float tot = 0.f;
#pragma unroll
  for (int i = 0; i < 8; ++i) tot += ps[i];
  float r = rsqrtf(tot * (1.0f / 1024.0f) + 1e-6f);
  float4 gv = *(const float4*)(g + tid * 4);
  unsigned short* orow = out + (size_t)row * 1024 + tid * 4;
  orow[0] = f2bfu(v.x * r * gv.x);
  orow[1] = f2bfu(v.y * r * gv.y);
  orow[2] = f2bfu(v.z * r * gv.z);
  orow[3] = f2bfu(v.w * r * gv.w);
}

// ---------------------------------------------------------------------------
// Precompute T5 relative-position bias table: table[h][rel + 2047].
// ---------------------------------------------------------------------------
__global__ void bias_table_kernel(const float* __restrict__ rel_bias,
                                  float* __restrict__ table) {
  int i = blockIdx.x * blockDim.x + threadIdx.x;
  if (i >= 16 * 4095) return;
  int h = i / 4095;
  int rel = (i % 4095) - 2047;           // mem - ctx
  int bucket = (rel > 0) ? 16 : 0;       // NUM_BUCKETS/2
  int ra = rel < 0 ? -rel : rel;
  if (ra < 8) {
    bucket += ra;
  } else {
    // 8 + log(ra/8)/log(16) * 8, clamped to 15
    int large = 8 + (int)(__logf((float)ra * 0.125f) * (8.0f / 2.7725887f));
    bucket += (large < 15) ? large : 15;
  }
  table[i] = rel_bias[bucket * 16 + h];
}

// ---------------------------------------------------------------------------
// Shared GEMM body: C[128x64] = A[128xD] @ W[Dx64] block tile, bf16 WMMA,
// double-buffered LDS, 2x2 tiles per wave. Epilogue differs per caller.
// ---------------------------------------------------------------------------
struct GemmAcc {
  v8f a[2][2];
};

__device__ __forceinline__ GemmAcc gemm_core(
    const unsigned short* __restrict__ A,   // [4096 x 1024] bf16 row-major
    const unsigned short* __restrict__ W,   // [1024 x 1024] bf16 row-major
    unsigned short (*sA)[128 * 40], unsigned short (*sB)[64 * 40],
    int row0, int col0) {
  constexpr int D = 1024, BK = 32, LDA = 40, LDB = 40;
  int tid = threadIdx.x;
  int wave = tid >> 5;
  int wm = wave >> 1;           // 0..3 -> 32-row slab
  int wn = wave & 1;            // 0..1 -> 32-col slab
  int lrow = tid >> 1, lcol = (tid & 1) * 16;   // A tile: 128x32
  int brow = tid >> 3, bcol = (tid & 7) * 8;    // W tile: 32x64

  GemmAcc acc;
#pragma unroll
  for (int i = 0; i < 2; ++i)
#pragma unroll
    for (int j = 0; j < 2; ++j)
      acc.a[i][j] = (v8f){0.f, 0.f, 0.f, 0.f, 0.f, 0.f, 0.f, 0.f};

  const unsigned short* aptr = A + (size_t)(row0 + lrow) * D + lcol;
  const unsigned short* wptr = W + (size_t)brow * 1024 + col0 + bcol;

  // Preload k0 = 0 into buffer 0.
  {
    uint4 a4a = *(const uint4*)(aptr);
    uint4 a4b = *(const uint4*)(aptr + 8);
    U8x16 wt;
    wt.u = *(const uint4*)(wptr);
    *(uint4*)(&sA[0][lrow * LDA + lcol])     = a4a;
    *(uint4*)(&sA[0][lrow * LDA + lcol + 8]) = a4b;
#pragma unroll
    for (int i = 0; i < 8; ++i) sB[0][(bcol + i) * LDB + brow] = wt.h[i];
  }
  __syncthreads();

  int buf = 0;
  for (int k0 = 0; k0 < D; k0 += BK) {
    bool has_next = (k0 + BK) < D;
    uint4 na = {}, nb = {};
    U8x16 nw; nw.u = (uint4){};
    if (has_next) {
      na = *(const uint4*)(aptr + k0 + BK);
      nb = *(const uint4*)(aptr + k0 + BK + 8);
      nw.u = *(const uint4*)(wptr + (size_t)(k0 + BK) * 1024);
      if (k0 + 2 * BK < D) {
        __builtin_prefetch(aptr + k0 + 2 * BK, 0, 1);
        __builtin_prefetch(wptr + (size_t)(k0 + 2 * BK) * 1024, 0, 1);
      }
    }

    v16bf a0 = load_frag_a(&sA[buf][(wm * 32) * LDA], LDA);
    v16bf a1 = load_frag_a(&sA[buf][(wm * 32 + 16) * LDA], LDA);
    v16bf b0 = load_frag_b(&sB[buf][(wn * 32) * LDB], LDB);
    v16bf b1 = load_frag_b(&sB[buf][(wn * 32 + 16) * LDB], LDB);
    acc.a[0][0] = WMMA_BF16(a0, b0, acc.a[0][0]);
    acc.a[0][1] = WMMA_BF16(a0, b1, acc.a[0][1]);
    acc.a[1][0] = WMMA_BF16(a1, b0, acc.a[1][0]);
    acc.a[1][1] = WMMA_BF16(a1, b1, acc.a[1][1]);

    if (has_next) {
      int nbuf = buf ^ 1;
      *(uint4*)(&sA[nbuf][lrow * LDA + lcol])     = na;
      *(uint4*)(&sA[nbuf][lrow * LDA + lcol + 8]) = nb;
#pragma unroll
      for (int i = 0; i < 8; ++i) sB[nbuf][(bcol + i) * LDB + brow] = nw.h[i];
      __syncthreads();
      buf = nbuf;
    }
  }
  return acc;
}

// ---------------------------------------------------------------------------
// QKV GEMM: normed @ W -> [B,H,S,64] bf16. blockIdx.z selects q/k/v.
// Grid (4096/128, 1024/64, 3) = (32, 16, 3).
// ---------------------------------------------------------------------------
__global__ __launch_bounds__(256) void qkv_gemm_kernel(
    const unsigned short* __restrict__ A,
    const unsigned short* __restrict__ Wq,
    const unsigned short* __restrict__ Wk,
    const unsigned short* __restrict__ Wv,
    unsigned short* __restrict__ Qo,
    unsigned short* __restrict__ Ko,
    unsigned short* __restrict__ Vo) {
  const unsigned short* W;
  unsigned short* O;
  if (blockIdx.z == 0)      { W = Wq; O = Qo; }
  else if (blockIdx.z == 1) { W = Wk; O = Ko; }
  else                      { W = Wv; O = Vo; }

  __shared__ unsigned short sA[2][128 * 40];
  __shared__ unsigned short sB[2][64 * 40];
  int row0 = blockIdx.x * 128, col0 = blockIdx.y * 64;

  GemmAcc acc = gemm_core(A, W, sA, sB, row0, col0);

  int tid = threadIdx.x, wave = tid >> 5, lane = tid & 31;
  int wm = wave >> 1, wn = wave & 1;
  int nn = lane & 15, mb = (lane < 16) ? 0 : 8;
#pragma unroll
  for (int i = 0; i < 2; ++i) {
#pragma unroll
    for (int j = 0; j < 2; ++j) {
#pragma unroll
      for (int r = 0; r < 8; ++r) {
        int row = row0 + wm * 32 + i * 16 + mb + r;
        int bb = row >> 11, ss = row & 2047;
        int col = col0 + wn * 32 + j * 16 + nn;
        O[(((size_t)bb * 16 + (col >> 6)) * 2048 + ss) * 64 + (col & 63)] =
            f2bfu(acc.a[i][j][r]);
      }
    }
  }
}

// ---------------------------------------------------------------------------
// Output projection + residual: out = ctx(bf16) @ wo(bf16) + hidden(fp32).
// Grid (32, 16).
// ---------------------------------------------------------------------------
__global__ __launch_bounds__(256) void out_gemm_kernel(
    const unsigned short* __restrict__ A,     // ctx [4096 x 1024] bf16
    const unsigned short* __restrict__ W,     // wo  [1024 x 1024] bf16
    const float* __restrict__ resid,          // hidden [4096 x 1024] fp32
    float* __restrict__ out) {
  __shared__ unsigned short sA[2][128 * 40];
  __shared__ unsigned short sB[2][64 * 40];
  int row0 = blockIdx.x * 128, col0 = blockIdx.y * 64;

  GemmAcc acc = gemm_core(A, W, sA, sB, row0, col0);

  int tid = threadIdx.x, wave = tid >> 5, lane = tid & 31;
  int wm = wave >> 1, wn = wave & 1;
  int nn = lane & 15, mb = (lane < 16) ? 0 : 8;
#pragma unroll
  for (int i = 0; i < 2; ++i) {
#pragma unroll
    for (int j = 0; j < 2; ++j) {
#pragma unroll
      for (int r = 0; r < 8; ++r) {
        int row = row0 + wm * 32 + i * 16 + mb + r;
        int col = col0 + wn * 32 + j * 16 + nn;
        out[(size_t)row * 1024 + col] =
            acc.a[i][j][r] + resid[(size_t)row * 1024 + col];
      }
    }
  }
}

// ---------------------------------------------------------------------------
// Flash attention: grid (S/128, H, B). Block = 8 waves; wave w owns q rows
// [qbase + 16w, +16). KV swept in 64-wide chunks staged in LDS.
// ---------------------------------------------------------------------------
__global__ __launch_bounds__(256) void attn_kernel(
    const unsigned short* __restrict__ Q,
    const unsigned short* __restrict__ K,
    const unsigned short* __restrict__ V,
    const float* __restrict__ btab_all,
    unsigned short* __restrict__ ctx) {
  constexpr int S = 2048, LDK = 72, LDP = 72;
  int h = blockIdx.y, b = blockIdx.z;
  int tid = threadIdx.x, wave = tid >> 5, lane = tid & 31;
  int nn = lane & 15, mb = (lane < 16) ? 0 : 8;
  size_t headoff = ((size_t)b * 16 + h) * S * 64;
  const unsigned short* Qh = Q + headoff;
  const unsigned short* Kh = K + headoff;
  const unsigned short* Vh = V + headoff;
  const float* bt = btab_all + h * 4095;

  __shared__ unsigned short sK[64 * LDK];        // [kv][d]
  __shared__ unsigned short sVt[64 * LDK];       // [d][kv]
  __shared__ unsigned short sP[8][16 * LDP];     // per-wave [m][kv]

  int qrow = blockIdx.x * 128 + wave * 16;
  v16bf qa0 = load_frag_a(Qh + (size_t)qrow * 64, 64);       // d = 0..31
  v16bf qa1 = load_frag_a(Qh + (size_t)qrow * 64 + 32, 64);  // d = 32..63

  float rowm[8], rowl[8];
  v8f o[4];
#pragma unroll
  for (int r = 0; r < 8; ++r) { rowm[r] = -3.0e38f; rowl[r] = 0.f; }
#pragma unroll
  for (int t = 0; t < 4; ++t) o[t] = (v8f){0.f, 0.f, 0.f, 0.f, 0.f, 0.f, 0.f, 0.f};

  int lrow = tid >> 2;          // 0..63
  int lcol = (tid & 3) * 16;    // 0,16,32,48

  for (int kv0 = 0; kv0 < S; kv0 += 64) {
    // Stage K chunk (row-major) and V chunk (transposed).
    *(uint4*)(&sK[lrow * LDK + lcol]) =
        *(const uint4*)(Kh + (size_t)(kv0 + lrow) * 64 + lcol);
    *(uint4*)(&sK[lrow * LDK + lcol + 8]) =
        *(const uint4*)(Kh + (size_t)(kv0 + lrow) * 64 + lcol + 8);
    U8x16 v4a, v4b;
    v4a.u = *(const uint4*)(Vh + (size_t)(kv0 + lrow) * 64 + lcol);
    v4b.u = *(const uint4*)(Vh + (size_t)(kv0 + lrow) * 64 + lcol + 8);
#pragma unroll
    for (int i = 0; i < 8; ++i) {
      sVt[(lcol + i) * LDK + lrow]     = v4a.h[i];
      sVt[(lcol + 8 + i) * LDK + lrow] = v4b.h[i];
    }
    if (kv0 + 64 < S) {
      __builtin_prefetch(Kh + (size_t)(kv0 + 64 + lrow) * 64 + lcol, 0, 1);
      __builtin_prefetch(Vh + (size_t)(kv0 + 64 + lrow) * 64 + lcol, 0, 1);
    }
    __syncthreads();

    // Scores: 16 q rows x 64 kv cols = 4 tiles, 2 k-steps (d=0..31, 32..63).
    v8f sc[4];
#pragma unroll
    for (int t = 0; t < 4; ++t) {
      v16bf bk0 = load_frag_b(&sK[(t * 16) * LDK], LDK);
      v16bf bk1 = load_frag_b(&sK[(t * 16) * LDK + 32], LDK);
      v8f a = {0.f, 0.f, 0.f, 0.f, 0.f, 0.f, 0.f, 0.f};
      a = WMMA_BF16(qa0, bk0, a);
      a = WMMA_BF16(qa1, bk1, a);
      sc[t] = a;
    }

    // Bias add + online softmax update, row-by-row.
#pragma unroll
    for (int r = 0; r < 8; ++r) {
      int q = qrow + mb + r;
      float mx = -3.0e38f;
#pragma unroll
      for (int t = 0; t < 4; ++t) {
        int kv = kv0 + t * 16 + nn;
        float x = sc[t][r] + bt[kv - q + 2047];
        sc[t][r] = x;
        mx = fmaxf(mx, x);
      }
#pragma unroll
      for (int off = 8; off >= 1; off >>= 1)
        mx = fmaxf(mx, __shfl_xor(mx, off, 32));
      float nm = fmaxf(rowm[r], mx);
      float corr = __expf(rowm[r] - nm);
      float rs = 0.f;
#pragma unroll
      for (int t = 0; t < 4; ++t) {
        float p = __expf(sc[t][r] - nm);
        sc[t][r] = p;
        rs += p;
      }
#pragma unroll
      for (int off = 8; off >= 1; off >>= 1)
        rs += __shfl_xor(rs, off, 32);
      rowl[r] = rowl[r] * corr + rs;
      rowm[r] = nm;
#pragma unroll
      for (int t = 0; t < 4; ++t) o[t][r] *= corr;
      // Spill P (C layout) into wave-private LDS as bf16 [m][kv].
#pragma unroll
      for (int t = 0; t < 4; ++t)
        sP[wave][(mb + r) * LDP + t * 16 + nn] = f2bfu(sc[t][r]);
    }

    // Re-read P as A-fragments (same-wave LDS: in-order, no barrier needed).
    v16bf pa0 = load_frag_a(&sP[wave][0], LDP);    // kv = 0..31
    v16bf pa1 = load_frag_a(&sP[wave][32], LDP);   // kv = 32..63

#pragma unroll
    for (int td = 0; td < 4; ++td) {
      v16bf bv0 = load_frag_b(&sVt[(td * 16) * LDK], LDK);
      v16bf bv1 = load_frag_b(&sVt[(td * 16) * LDK + 32], LDK);
      o[td] = WMMA_BF16(pa0, bv0, o[td]);
      o[td] = WMMA_BF16(pa1, bv1, o[td]);
    }
    __syncthreads();
  }

  // Normalize and write ctx in [B,S,H*64] layout (bf16) for the final GEMM.
#pragma unroll
  for (int r = 0; r < 8; ++r) {
    float inv = 1.0f / rowl[r];
    int q = qrow + mb + r;
    size_t base = ((size_t)b * 2048 + q) * 1024 + h * 64 + nn;
#pragma unroll
    for (int td = 0; td < 4; ++td)
      ctx[base + td * 16] = f2bfu(o[td][r] * inv);
  }
}

// ---------------------------------------------------------------------------
// Host launcher. Inputs: hidden, ln_weight, wq, wk, wv, wo, rel_bias.
// Workspace layout (~49 MB): bf16 weights, normed, q/k/v, ctx, bias table.
// ---------------------------------------------------------------------------
extern "C" void kernel_launch(void* const* d_in, const int* in_sizes, int n_in,
                              void* d_out, int out_size, void* d_ws, size_t ws_size,
                              hipStream_t stream) {
  (void)in_sizes; (void)n_in; (void)out_size; (void)ws_size;
  const float* hidden = (const float*)d_in[0];
  const float* lnw    = (const float*)d_in[1];
  const float* wq     = (const float*)d_in[2];
  const float* wk     = (const float*)d_in[3];
  const float* wv     = (const float*)d_in[4];
  const float* wo     = (const float*)d_in[5];
  const float* rb     = (const float*)d_in[6];
  float* out = (float*)d_out;

  char* ws = (char*)d_ws;
  size_t off = 0;
  auto take = [&](size_t bytes) -> char* {
    char* p = ws + off;
    off += (bytes + 255) & ~(size_t)255;
    return p;
  };
  const size_t WN = 1024ull * 1024;       // weight elements
  const size_t AN = 4096ull * 1024;       // activation elements (B*S x D)
  unsigned short* wq_b   = (unsigned short*)take(WN * 2);
  unsigned short* wk_b   = (unsigned short*)take(WN * 2);
  unsigned short* wv_b   = (unsigned short*)take(WN * 2);
  unsigned short* wo_b   = (unsigned short*)take(WN * 2);
  unsigned short* normed = (unsigned short*)take(AN * 2);
  unsigned short* qb     = (unsigned short*)take(AN * 2);
  unsigned short* kb     = (unsigned short*)take(AN * 2);
  unsigned short* vb     = (unsigned short*)take(AN * 2);
  unsigned short* ctxb   = (unsigned short*)take(AN * 2);
  float* btab            = (float*)take(16ull * 4095 * 4);

  cvt_bf16_kernel<<<(int)(WN / 256), 256, 0, stream>>>(wq, wq_b, (int)WN);
  cvt_bf16_kernel<<<(int)(WN / 256), 256, 0, stream>>>(wk, wk_b, (int)WN);
  cvt_bf16_kernel<<<(int)(WN / 256), 256, 0, stream>>>(wv, wv_b, (int)WN);
  cvt_bf16_kernel<<<(int)(WN / 256), 256, 0, stream>>>(wo, wo_b, (int)WN);
  rmsnorm_kernel<<<4096, 256, 0, stream>>>(hidden, lnw, normed);
  bias_table_kernel<<<(16 * 4095 + 255) / 256, 256, 0, stream>>>(rb, btab);
  qkv_gemm_kernel<<<dim3(32, 16, 3), 256, 0, stream>>>(normed, wq_b, wk_b, wv_b,
                                                       qb, kb, vb);
  attn_kernel<<<dim3(16, 16, 2), 256, 0, stream>>>(qb, kb, vb, btab, ctxb);
  out_gemm_kernel<<<dim3(32, 16), 256, 0, stream>>>(ctxb, wo_b, hidden, out);
}
